// RGAT_5789615915678
// MI455X (gfx1250) — compile-verified, matrix operand order checked by reading, not run on hardware
//
#include <hip/hip_runtime.h>
#include <hip/hip_fp16.h>

// ---------------------------------------------------------------------------
// RGAT forward for gfx1250 (MI455X).
//  - Dense projections: v_wmma_f32_16x16x32_f16; one wave = 32x64 output strip
//    (2 M-tiles x 4 N-tiles -> 1.5 b128 loads per WMMA), no guards inside the
//    K-loop so EXEC stays all-ones and loads pipeline with WMMA.
//  - Edge softmax/aggregate: fp32 logits/z, packed-f16 atomics
//    (global_atomic_pk_add_f16) for the dominant message scatter.
// ---------------------------------------------------------------------------

typedef _Float16 v16h __attribute__((ext_vector_type(16)));
typedef _Float16 v8h  __attribute__((ext_vector_type(8)));
typedef float    v8f  __attribute__((ext_vector_type(8)));

#define RGAT_N   100000
#define RGAT_E   600000
#define RGAT_R   3
#define RGAT_HF  256
#define RGAT_H   4
#define RGAT_D   64
#define RGAT_C   153
#define GK       256          // K dim of every GEMM here (IN == HF == 256)

// ------------------------------ small kernels ------------------------------

__global__ void k_cvt_f16(const float* __restrict__ in, _Float16* __restrict__ out,
                          int n, int relu) {
  int t = blockIdx.x * blockDim.x + threadIdx.x;
  if (t >= n) return;
  float v = in[t];
  if (relu) v = fmaxf(v, 0.0f);
  out[t] = (_Float16)v;
}

// W [K x ncols] (row-major, f32)  ->  Wt [npad x K] (f16), zero-padded rows.
__global__ void k_tcvt(const float* __restrict__ W, _Float16* __restrict__ Wt,
                       int K, int ncols, int total /* = npad*K */) {
  int t = blockIdx.x * blockDim.x + threadIdx.x;
  if (t >= total) return;
  int nrow = t / K;
  int k    = t - nrow * K;
  float v = (nrow < ncols) ? W[(size_t)k * ncols + nrow] : 0.0f;
  Wt[(size_t)nrow * K + k] = (_Float16)v;
}

__global__ void k_fill(float* __restrict__ p, float v, int n) {
  int t = blockIdx.x * blockDim.x + threadIdx.x;
  if (t < n) p[t] = v;
}

// acc[n*256+j] = sum_r b[r*256+j]   (per-node bias, summed over relations)
__global__ void k_bias_init(float* __restrict__ acc, const float* __restrict__ b,
                            int total) {
  int t = blockIdx.x * blockDim.x + threadIdx.x;
  if (t >= total) return;
  int j = t & (RGAT_HF - 1);
  acc[t] = b[j] + b[RGAT_HF + j] + b[2 * RGAT_HF + j];
}

// ------------------------------ WMMA GEMM ----------------------------------
// C[M x ncols] (f32) = A[M x 256] (f16 row-major) * Bt[n][k] (f16) (+ bias)
// One wave computes a 32x(16*NSUB) strip: 2 M-tiles x NSUB N-tiles.
// No conditionals in the K-loop -> EXEC all-ones, loads pipeline with WMMA.

__device__ inline v16h ldfrag(const _Float16* p) {
  v8h lo = *(const v8h*)(p);        // K = kb .. kb+7
  v8h hi = *(const v8h*)(p + 16);   // K = kb+16 .. kb+23
  return __builtin_shufflevector(lo, hi, 0,1,2,3,4,5,6,7,8,9,10,11,12,13,14,15);
}

template <int NSUB>
__global__ void k_wmma_gemm(const _Float16* __restrict__ A,
                            const _Float16* __restrict__ Bt,
                            float* __restrict__ C,
                            const float* __restrict__ bias,
                            int mgroups, int ngroups, int tile0, int ncols, int ldc) {
  int wave = (blockIdx.x * blockDim.x + threadIdx.x) >> 5;  // uniform per wave
  int lane = threadIdx.x & 31;
  if (wave >= mgroups * ngroups) return;                    // whole-wave exit
  int mg  = wave / ngroups;
  int nt0 = tile0 + (wave - mg * ngroups) * NSUB;

  int mrow0 = (mg << 5) + (lane & 15);      // first of two A row-tiles
  int kb    = (lane >> 4) << 3;             // 0 (lanes 0-15) or 8 (lanes 16-31)

  const _Float16* pa0 = A + (size_t)mrow0 * GK + kb;
  const _Float16* pa1 = pa0 + (size_t)16 * GK;
  const _Float16* pb  = Bt + (size_t)((nt0 << 4) + (lane & 15)) * GK + kb;

  v8f c0[NSUB] = {};
  v8f c1[NSUB] = {};
#pragma unroll
  for (int kk = 0; kk < GK; kk += 32) {
    __builtin_prefetch(pa0 + kk + 128, 0, 1);       // global_prefetch_b8
    __builtin_prefetch(pa1 + kk + 128, 0, 1);
    v16h av0 = ldfrag(pa0 + kk);
    v16h av1 = ldfrag(pa1 + kk);
#pragma unroll
    for (int i = 0; i < NSUB; ++i) {
      const _Float16* pbi = pb + (size_t)(i << 4) * GK + kk;
      __builtin_prefetch(pbi + 128, 0, 1);
      v16h bv = ldfrag(pbi);
      c0[i] = __builtin_amdgcn_wmma_f32_16x16x32_f16(false, av0, false, bv,
                                                     (short)0, c0[i], false, false);
      c1[i] = __builtin_amdgcn_wmma_f32_16x16x32_f16(false, av1, false, bv,
                                                     (short)0, c1[i], false, false);
    }
  }

  // C/D layout: lanes 0-15 hold M = base + j, lanes 16-31 hold M = base + 8 + j
  int r0 = (mg << 5) + ((lane >> 4) << 3);
#pragma unroll
  for (int i = 0; i < NSUB; ++i) {
    int ncol = ((nt0 + i) << 4) + (lane & 15);
    if (ncol < ncols) {                       // store-time guard only
      float bb = bias ? bias[ncol] : 0.0f;
#pragma unroll
      for (int j = 0; j < 8; ++j) {
        C[(size_t)(r0 + j) * ldc + ncol]      = c0[i][j] + bb;
        C[(size_t)(r0 + 16 + j) * ldc + ncol] = c1[i][j] + bb;
      }
    }
  }
}

// --------------------------- attention kernels -----------------------------

// el[n,h] = <feat[n,h,:], al[h,:]> ; er likewise.   One thread per (n,h).
__global__ void k_el_er(const float* __restrict__ feat,
                        const float* __restrict__ al,
                        const float* __restrict__ ar,
                        float* __restrict__ el, float* __restrict__ er, int NH) {
  int t = blockIdx.x * blockDim.x + threadIdx.x;
  if (t >= NH) return;
  int n = t >> 2, h = t & 3;
  const float4* f  = (const float4*)(feat + (size_t)n * RGAT_HF + h * RGAT_D);
  const float4* a4 = (const float4*)(al + h * RGAT_D);
  const float4* b4 = (const float4*)(ar + h * RGAT_D);
  float s1 = 0.f, s2 = 0.f;
#pragma unroll
  for (int i = 0; i < RGAT_D / 4; ++i) {
    float4 v = f[i], a = a4[i], b = b4[i];
    s1 += v.x * a.x + v.y * a.y + v.z * a.z + v.w * a.w;
    s2 += v.x * b.x + v.y * b.y + v.z * b.z + v.w * b.w;
  }
  el[t] = s1;
  er[t] = s2;
}

__device__ inline void atomicMaxF(float* addr, float val) {
  unsigned int* ua = reinterpret_cast<unsigned int*>(addr);
  unsigned int old = *ua;
  while (__uint_as_float(old) < val) {
    unsigned int assumed = old;
    old = atomicCAS(ua, assumed, __float_as_uint(val));
    if (old == assumed) break;
  }
}

__device__ inline float lrelu(float v) { return v > 0.f ? v : 0.2f * v; }

// pass A: e = leaky_relu(el[src]+er[dst]); cache e; segment max per (dst,head)
__global__ void k_edge_max(const int* __restrict__ sr, const int* __restrict__ dr,
                           const float* __restrict__ el, const float* __restrict__ er,
                           float* __restrict__ mx, float* __restrict__ ebuf, int EH) {
  int t = blockIdx.x * blockDim.x + threadIdx.x;
  if (t >= EH) return;
  int e = t >> 2, h = t & 3;
  int s = sr[e], d = dr[e];
  float v = lrelu(el[(s << 2) + h] + er[(d << 2) + h]);
  ebuf[t] = v;
  atomicMaxF(&mx[(d << 2) + h], v);
}

// pass B: a = exp(e - mx);  z[dst,h] += a;  S[dst,h,:] += a * feat[src,h,:]
// S accumulated as packed f16 (global_atomic_pk_add_f16) to halve traffic.
__global__ void k_edge_acc(const int* __restrict__ sr, const int* __restrict__ dr,
                           const float* __restrict__ ebuf,
                           const float* __restrict__ mx, float* __restrict__ z,
                           const float* __restrict__ feat, __half2* __restrict__ S,
                           int EH) {
  int t = blockIdx.x * blockDim.x + threadIdx.x;
  if (t >= EH) return;
  int e = t >> 2, h = t & 3;
  int s = sr[e], d = dr[e];
  float a = __expf(ebuf[t] - mx[(d << 2) + h]);
  atomicAdd(&z[(d << 2) + h], a);
  const float4* fp = reinterpret_cast<const float4*>(feat + (size_t)s * RGAT_HF + h * RGAT_D);
  __half2* Sp = S + (size_t)d * (RGAT_HF / 2) + h * (RGAT_D / 2);
#pragma unroll
  for (int i = 0; i < RGAT_D / 4; ++i) {
    float4 f = fp[i];
    atomicAdd(&Sp[2 * i + 0], __floats2half2_rn(a * f.x, a * f.y));
    atomicAdd(&Sp[2 * i + 1], __floats2half2_rn(a * f.z, a * f.w));
  }
}

// acc += S / z   (softmax normalization folded into the aggregate)
__global__ void k_combine(float* __restrict__ acc, const __half2* __restrict__ S,
                          const float* __restrict__ z, int totalPairs) {
  int t = blockIdx.x * blockDim.x + threadIdx.x;
  if (t >= totalPairs) return;
  int idx = t << 1;                       // first float element
  int n = idx >> 8;
  int h = (idx >> 6) & 3;
  float zz = z[(n << 2) + h];
  if (zz > 0.f) {
    float2 s = __half22float2(S[t]);
    acc[idx]     += s.x / zz;
    acc[idx + 1] += s.y / zz;
  }
}

// ------------------------------- host side ---------------------------------

extern "C" void kernel_launch(void* const* d_in, const int* in_sizes, int n_in,
                              void* d_out, int out_size, void* d_ws, size_t ws_size,
                              hipStream_t stream) {
  (void)in_sizes; (void)n_in; (void)out_size; (void)ws_size;
  const float* x    = (const float*)d_in[0];
  const int*   src  = (const int*)d_in[1];
  const int*   dst  = (const int*)d_in[2];
  const float* W1   = (const float*)d_in[3];
  const float* al1  = (const float*)d_in[4];
  const float* ar1  = (const float*)d_in[5];
  const float* b1   = (const float*)d_in[6];
  const float* W2   = (const float*)d_in[7];
  const float* al2  = (const float*)d_in[8];
  const float* ar2  = (const float*)d_in[9];
  const float* b2   = (const float*)d_in[10];
  const float* Wout = (const float*)d_in[11];
  const float* bout = (const float*)d_in[12];
  float* out = (float*)d_out;

  char* ws = (char*)d_ws;
  size_t off = 0;
  auto carve = [&](size_t bytes) -> char* {
    char* p = ws + off;
    off += (bytes + 255) & ~(size_t)255;
    return p;
  };
  _Float16* hf16 = (_Float16*)carve((size_t)RGAT_N * RGAT_HF * 2);   // 51.2 MB
  _Float16* Wt   = (_Float16*)carve((size_t)256 * 256 * 2);          // per-rel weight (reused)
  float* feat = (float*)carve((size_t)RGAT_N * RGAT_HF * 4);         // 102.4 MB
  float* bufA = (float*)carve((size_t)RGAT_N * RGAT_HF * 4);         // 102.4 MB
  float* bufB = (float*)carve((size_t)RGAT_N * RGAT_HF * 4);         // 102.4 MB
  float* el   = (float*)carve((size_t)RGAT_N * RGAT_H * 4);
  float* er   = (float*)carve((size_t)RGAT_N * RGAT_H * 4);
  float* mx   = (float*)carve((size_t)RGAT_N * RGAT_H * 4);
  float* z    = (float*)carve((size_t)RGAT_N * RGAT_H * 4);
  float* ebuf = (float*)carve((size_t)RGAT_E * RGAT_H * 4);          // 9.6 MB

  const int NHF = RGAT_N * RGAT_HF;   // 25.6M
  const int NH  = RGAT_N * RGAT_H;    // 400k
  const int EH  = RGAT_E * RGAT_H;    // 2.4M
  const int MG  = RGAT_N / 32;        // 3125 row-tile pairs (exact)
  dim3 blk(256);
  auto g = [](int n) { return dim3((unsigned)((n + 255) / 256)); };

  // GEMM over all M rows; ntiles column tiles of 16.  Full groups of 4 plus
  // an optional NSUB=2 remainder launch (153-col output case: 10 = 2*4 + 2).
  auto launch_gemm = [&](const _Float16* A, const _Float16* B, float* C,
                         const float* bias, int ntiles, int ncols, int ldc) {
    int full = ntiles >> 2;             // groups of 4 tiles
    int rem  = ntiles & 3;              // 0 or 2 for our shapes
    if (full > 0) {
      int threads = MG * full * 32;
      k_wmma_gemm<4><<<g(threads), blk, 0, stream>>>(A, B, C, bias, MG, full,
                                                     0, ncols, ldc);
    }
    if (rem == 2) {
      int threads = MG * 32;
      k_wmma_gemm<2><<<g(threads), blk, 0, stream>>>(A, B, C, bias, MG, 1,
                                                     full << 2, ncols, ldc);
    }
  };

  auto run_layer = [&](const float* Wl, const float* all, const float* arl,
                       const float* bl, float* acc, float* Sraw) {
    __half2* S = (__half2*)Sraw;        // aliases the idle ping-pong buffer
    k_bias_init<<<g(NHF), blk, 0, stream>>>(acc, bl, NHF);
    for (int r = 0; r < RGAT_R; ++r) {
      k_tcvt<<<g(256 * 256), blk, 0, stream>>>(Wl + (size_t)r * 256 * 256, Wt,
                                               GK, 256, 256 * 256);
      launch_gemm(hf16, Wt, feat, nullptr, RGAT_HF / 16, RGAT_HF, RGAT_HF);
      k_el_er<<<g(NH), blk, 0, stream>>>(feat, all + r * RGAT_H * RGAT_D,
                                         arl + r * RGAT_H * RGAT_D, el, er, NH);
      k_fill<<<g(NH), blk, 0, stream>>>(mx, -3.0e38f, NH);
      k_fill<<<g(NH), blk, 0, stream>>>(z, 0.0f, NH);
      k_fill<<<g(NHF / 2), blk, 0, stream>>>((float*)S, 0.0f, NHF / 2); // zero f16 S
      k_edge_max<<<g(EH), blk, 0, stream>>>(src + (size_t)r * RGAT_E,
                                            dst + (size_t)r * RGAT_E, el, er, mx,
                                            ebuf, EH);
      k_edge_acc<<<g(EH), blk, 0, stream>>>(src + (size_t)r * RGAT_E,
                                            dst + (size_t)r * RGAT_E, ebuf, mx, z,
                                            feat, S, EH);
      k_combine<<<g(NHF / 2), blk, 0, stream>>>(acc, S, z, NHF / 2);
    }
  };

  // ---- layer 1: h = RGATLayer(x) ----
  k_cvt_f16<<<g(NHF), blk, 0, stream>>>(x, hf16, NHF, 0);
  run_layer(W1, al1, ar1, b1, /*acc=*/bufA, /*S=*/bufB);
  // relu + downconvert for layer 2
  k_cvt_f16<<<g(NHF), blk, 0, stream>>>(bufA, hf16, NHF, 1);

  // ---- layer 2 ----
  run_layer(W2, al2, ar2, b2, /*acc=*/bufB, /*S=*/bufA);
  k_cvt_f16<<<g(NHF), blk, 0, stream>>>(bufB, hf16, NHF, 0);

  // ---- output linear: [N,256] @ [256,153] + bout ----
  k_tcvt<<<g(160 * 256), blk, 0, stream>>>(Wout, Wt, GK, RGAT_C, 160 * 256);
  launch_gemm(hf16, Wt, out, bout, 160 / 16, RGAT_C, RGAT_C);
}
